// FPModule_94489280936
// MI455X (gfx1250) — matrix operand order repaired and checked by reading, not compiled, not run on hardware
//
#include <hip/hip_runtime.h>
#include <hip/hip_bf16.h>

// ---------- problem constants ----------
#define BB      8
#define N_SRC   2048
#define N_DST   8192
#define M_ROWS  (BB * N_DST)          // 65536
#define C_SRC   128
#define C_SKIP  64
#define C_IN    (C_SRC + C_SKIP)      // 192
#define C_H     128
#define LRELU   0.01f
#define BN_EPS  1e-5f

typedef _Float16 v16h __attribute__((ext_vector_type(16)));
typedef _Float16 v8h  __attribute__((ext_vector_type(8)));
typedef _Float16 v4h  __attribute__((ext_vector_type(4)));
typedef float    v8f  __attribute__((ext_vector_type(8)));
typedef unsigned int v4u __attribute__((ext_vector_type(4)));
typedef int      v8i  __attribute__((ext_vector_type(8)));
typedef int      v4i  __attribute__((ext_vector_type(4)));

// ---------- workspace layout (bytes) ----------
static constexpr size_t H0_OFF    = 0;                                  // f16 [65536,192]
static constexpr size_t H1_OFF    = H0_OFF + (size_t)M_ROWS * C_IN * 2; // f16 [65536,128]
static constexpr size_t H2_OFF    = H1_OFF + (size_t)M_ROWS * C_H * 2;  // f32 [65536,128]
static constexpr size_t W1H_OFF   = H2_OFF + (size_t)M_ROWS * C_H * 4;  // f16 [128,192]
static constexpr size_t W2H_OFF   = W1H_OFF + (size_t)C_H * C_IN * 2;   // f16 [128,128]
static constexpr size_t STATS_OFF = W2H_OFF + (size_t)C_H * C_H * 2;    // f32 sum1,sq1,sum2,sq2 (4*128)
static constexpr size_t S2_OFF    = STATS_OFF + 512 * 4;                // f32 [128]
static constexpr size_t T2_OFF    = S2_OFF + 128 * 4;                   // f32 [128]
static constexpr size_t B2P_OFF   = T2_OFF + 128 * 4;                   // f32 [128]

// output layout (floats)
static constexpr size_t OUT_H   = (size_t)M_ROWS * C_H;     // 8388608
static constexpr size_t OUT_POS = OUT_H;                    // pos_skip start
static constexpr size_t OUT_BAT = OUT_H + (size_t)M_ROWS*3; // batch start

// ---------------------------------------------------------------
// 0) prep: zero BN stat accumulators, convert W1 -> f16
// ---------------------------------------------------------------
__global__ __launch_bounds__(256) void prep_kernel(const float* __restrict__ W1,
                                                   _Float16* __restrict__ w1h,
                                                   float* __restrict__ stats) {
  int tid = blockIdx.x * 256 + threadIdx.x;
  if (tid < 512) stats[tid] = 0.0f;
  for (int i = tid; i < C_H * C_IN; i += gridDim.x * 256)
    w1h[i] = (_Float16)W1[i];
}

// ---------------------------------------------------------------
// 1) kNN(K=3) + inverse-d^2 interpolation + concat -> h0 (f16)
//    one thread per dst point; src positions staged through LDS
// ---------------------------------------------------------------
__global__ __launch_bounds__(256) void knn_interp_kernel(const float* __restrict__ x,
                                                         const float* __restrict__ pos,
                                                         const float* __restrict__ x_skip,
                                                         const float* __restrict__ pos_skip,
                                                         _Float16* __restrict__ h0) {
  __shared__ float sp[256 * 3];
  const int tid = threadIdx.x;
  const int b   = blockIdx.x >> 5;      // 32 blocks per cloud
  const int t   = blockIdx.x & 31;
  const int dst = (b << 13) + (t << 8) + tid;

  const float px = pos_skip[dst * 3 + 0];
  const float py = pos_skip[dst * 3 + 1];
  const float pz = pos_skip[dst * 3 + 2];

  float d0 = 3.4e38f, d1 = 3.4e38f, d2 = 3.4e38f;
  int   i0 = 0, i1 = 0, i2 = 0;
  const int srcBase = b << 11;          // b * 2048

  for (int c0 = 0; c0 < N_SRC; c0 += 256) {
    __syncthreads();
    const int s = srcBase + c0 + tid;
    sp[tid * 3 + 0] = pos[s * 3 + 0];
    sp[tid * 3 + 1] = pos[s * 3 + 1];
    sp[tid * 3 + 2] = pos[s * 3 + 2];
    __syncthreads();
    #pragma unroll 8
    for (int j = 0; j < 256; ++j) {
      const float dx = px - sp[j * 3 + 0];
      const float dy = py - sp[j * 3 + 1];
      const float dz = pz - sp[j * 3 + 2];
      const float d  = dx * dx + dy * dy + dz * dz;
      const int  id  = c0 + j;
      if (d < d2) {
        if (d < d1) {
          d2 = d1; i2 = i1;
          if (d < d0) { d1 = d0; i1 = i0; d0 = d; i0 = id; }
          else        { d1 = d;  i1 = id; }
        } else { d2 = d; i2 = id; }
      }
    }
  }

  float w0 = 1.0f / fmaxf(d0, 1e-16f);
  float w1 = 1.0f / fmaxf(d1, 1e-16f);
  float w2 = 1.0f / fmaxf(d2, 1e-16f);
  const float inv = 1.0f / (w0 + w1 + w2);
  w0 *= inv; w1 *= inv; w2 *= inv;

  const float4* f0 = (const float4*)(x + (size_t)(srcBase + i0) * C_SRC);
  const float4* f1 = (const float4*)(x + (size_t)(srcBase + i1) * C_SRC);
  const float4* f2 = (const float4*)(x + (size_t)(srcBase + i2) * C_SRC);
  _Float16* hrow = h0 + (size_t)dst * C_IN;

  #pragma unroll 4
  for (int c = 0; c < C_SRC / 4; ++c) {
    const float4 a = f0[c], bq = f1[c], cq = f2[c];
    v4h o = { (_Float16)(w0 * a.x + w1 * bq.x + w2 * cq.x),
              (_Float16)(w0 * a.y + w1 * bq.y + w2 * cq.y),
              (_Float16)(w0 * a.z + w1 * bq.z + w2 * cq.z),
              (_Float16)(w0 * a.w + w1 * bq.w + w2 * cq.w) };
    *(v4h*)(hrow + c * 4) = o;
  }
  const float4* xs = (const float4*)(x_skip + (size_t)dst * C_SKIP);
  #pragma unroll 4
  for (int c = 0; c < C_SKIP / 4; ++c) {
    const float4 a = xs[c];
    v4h o = { (_Float16)a.x, (_Float16)a.y, (_Float16)a.z, (_Float16)a.w };
    *(v4h*)(hrow + C_SRC + c * 4) = o;
  }
}

// ---------------------------------------------------------------
// 2/4) WMMA GEMM: Y = lrelu(A[M,KTOT] * W^T + bias), W in [128,KTOT]
//      per-channel sum/sumsq accumulated for BatchNorm.
//      Weights staged into LDS by the Tensor Data Mover (D# per ISA ch.8);
//      each wave computes 16 rows x 128 cols with A fragments preloaded.
// ---------------------------------------------------------------
template <int KTOT, bool OUT_F16>
__global__ __launch_bounds__(256) void gemm_bn_kernel(const _Float16* __restrict__ A,
                                                      const _Float16* __restrict__ Wg,
                                                      const float* __restrict__ bias,
                                                      _Float16* __restrict__ out16,
                                                      float* __restrict__ out32,
                                                      float* __restrict__ gSum,
                                                      float* __restrict__ gSq) {
  constexpr int KS = KTOT / 32;
  __shared__ _Float16 sW[C_H * KTOT];
  __shared__ float sSum[C_H], sSq[C_H];

  const int tid = threadIdx.x;
  if (tid < C_H) { sSum[tid] = 0.0f; sSq[tid] = 0.0f; }

#if __has_builtin(__builtin_amdgcn_tensor_load_to_lds)
  if (tid < 32) {
    // Tensor DMA: 2D tile [128 x KTOT] of 2-byte elements, global -> LDS.
    const unsigned long long ga = (unsigned long long)(uintptr_t)Wg;
    const unsigned ldsOff = (unsigned)(uintptr_t)(void*)sW; // addr[31:0] = LDS offset
    v4u g0;
    g0[0] = 1u;                                           // count=1 (valid user D#)
    g0[1] = ldsOff;                                       // lds_addr
    g0[2] = (unsigned)(ga & 0xFFFFFFFFu);                 // global_addr[31:0]
    g0[3] = (unsigned)((ga >> 32) & 0x1FFFFFFu)           // global_addr[56:32]
          | (2u << 30);                                   // type=2 ("image")
    v8i g1;
    g1[0] = 0x00010000;                     // wg_mask=0, data_size=1 (2 bytes)
    g1[1] = (int)(((unsigned)KTOT & 0xFFFFu) << 16);   // tensor_dim0[15:0]
    g1[2] = (int)((unsigned)C_H << 16);     // tensor_dim0 hi=0 | tensor_dim1 lo=128
    g1[3] = (int)((unsigned)KTOT << 16);    // tensor_dim1 hi=0 | tile_dim0=KTOT
    g1[4] = C_H;                            // tile_dim1=128, tile_dim2=0
    g1[5] = KTOT;                           // tensor_dim0_stride low
    g1[6] = 0;
    g1[7] = 0;
    v4i gz  = {0, 0, 0, 0};
    v8i gz8 = {0, 0, 0, 0, 0, 0, 0, 0};
    __builtin_amdgcn_tensor_load_to_lds(g0, g1, gz, gz, gz8, 0);
    __builtin_amdgcn_s_wait_tensorcnt((short)0);
  }
#else
  { // fallback: cooperative weight stage global -> LDS (b128 chunks)
    const v8h* src = (const v8h*)Wg;
    v8h* dst = (v8h*)sW;
    for (int i = tid; i < C_H * KTOT / 8; i += 256) dst[i] = src[i];
  }
#endif
  __syncthreads();

  const int wave  = tid >> 5;
  const int lane  = tid & 31;
  const int laneM = lane & 15;
  const int hi    = lane >> 4;
  const int rowBase = (blockIdx.x * 8 + wave) * 16;

  union U16 { v16h v; v8h h[2]; };

  // Preload A fragments: 16-bit A 16x32 layout (ISA 7.12.2):
  //   lanes 0-15 hold K {0..7, 16..23}; lanes 16-31 hold K {8..15, 24..31}
  const _Float16* Arow = A + (size_t)(rowBase + laneM) * KTOT;
  v16h afr[KS];
  #pragma unroll
  for (int ks = 0; ks < KS; ++ks) {
    U16 u;
    u.h[0] = *(const v8h*)(Arow + ks * 32 + hi * 8);
    u.h[1] = *(const v8h*)(Arow + ks * 32 + 16 + hi * 8);
    afr[ks] = u.v;
  }

  #pragma unroll
  for (int n0 = 0; n0 < C_H / 16; ++n0) {
    const int col = n0 * 16 + laneM;
    // B 32x16 layout: lane n holds column n; lanes 0-15: K {0..7, 16..23}, lanes 16-31: K {8..15, 24..31}
    const _Float16* wrow = sW + (size_t)col * KTOT + hi * 8;
    v8f acc = {0.f, 0.f, 0.f, 0.f, 0.f, 0.f, 0.f, 0.f};
    #pragma unroll
    for (int ks = 0; ks < KS; ++ks) {
      U16 bfrag;
      bfrag.h[0] = *(const v8h*)(wrow + ks * 32);
      bfrag.h[1] = *(const v8h*)(wrow + ks * 32 + 16);
      acc = __builtin_amdgcn_wmma_f32_16x16x32_f16(
          /*neg_a=*/false, afr[ks], /*neg_b=*/false, bfrag.v,
          /*c_mod=*/(short)0, acc, /*reuse_a=*/false, /*reuse_b=*/false);
    }
    const float bv = bias[col];
    float s = 0.0f, q = 0.0f;
    #pragma unroll
    for (int v = 0; v < 8; ++v) {
      float y = acc[v] + bv;
      y = y > 0.0f ? y : LRELU * y;        // LeakyReLU
      s += y; q += y * y;
      const size_t o = (size_t)(rowBase + hi * 8 + v) * C_H + col;
      if constexpr (OUT_F16) out16[o] = (_Float16)y;
      else                   out32[o] = y;
    }
    atomicAdd(&sSum[col], s);
    atomicAdd(&sSq[col], q);
  }
  __syncthreads();
  if (tid < C_H) {
    atomicAdd(&gSum[tid], sSum[tid]);
    atomicAdd(&gSq[tid], sSq[tid]);
  }
}

// ---------------------------------------------------------------
// 3) BN1 finalize + fold into W2/b2:  W2' = W2 .* s1 (per-col),
//    b2' = b2 + W2 @ t1, where s1 = g1*rsqrt(var+eps), t1 = be1 - mean*s1
// ---------------------------------------------------------------
__global__ __launch_bounds__(128) void fold_kernel(const float* __restrict__ sum1,
                                                   const float* __restrict__ sq1,
                                                   const float* __restrict__ g1,
                                                   const float* __restrict__ be1,
                                                   const float* __restrict__ W2,
                                                   const float* __restrict__ b2,
                                                   _Float16* __restrict__ w2h,
                                                   float* __restrict__ b2p) {
  __shared__ float sS[C_H], sT[C_H];
  const int t = threadIdx.x;
  const float invM = 1.0f / (float)M_ROWS;
  const float m = sum1[t] * invM;
  const float v = sq1[t] * invM - m * m;
  const float s = g1[t] * rsqrtf(v + BN_EPS);
  sS[t] = s;
  sT[t] = be1[t] - m * s;
  __syncthreads();
  float acc = b2[t];
  for (int j = 0; j < C_H; ++j) {
    const float w = W2[t * C_H + j];
    acc += w * sT[j];
    w2h[t * C_H + j] = (_Float16)(w * sS[j]);
  }
  b2p[t] = acc;
}

// ---------------------------------------------------------------
// 5) BN2 finalize: per-channel scale/shift
// ---------------------------------------------------------------
__global__ __launch_bounds__(128) void finalize2_kernel(const float* __restrict__ sum2,
                                                        const float* __restrict__ sq2,
                                                        const float* __restrict__ g2,
                                                        const float* __restrict__ be2,
                                                        float* __restrict__ s2,
                                                        float* __restrict__ t2) {
  const int t = threadIdx.x;
  const float invM = 1.0f / (float)M_ROWS;
  const float m = sum2[t] * invM;
  const float v = sq2[t] * invM - m * m;
  const float s = g2[t] * rsqrtf(v + BN_EPS);
  s2[t] = s;
  t2[t] = be2[t] - m * s;
}

// ---------------------------------------------------------------
// 6) epilogue: out = h2*s2+t2 ; passthrough pos_skip ; batch = row/8192
// ---------------------------------------------------------------
__global__ __launch_bounds__(256) void epilogue_kernel(const float* __restrict__ h2,
                                                       const float* __restrict__ s2,
                                                       const float* __restrict__ t2,
                                                       const float* __restrict__ pos_skip,
                                                       float* __restrict__ out) {
  const int idx = blockIdx.x * 256 + threadIdx.x;
  const int row = idx >> 7;
  const int c   = idx & (C_H - 1);
  out[idx] = h2[idx] * s2[c] + t2[c];
  if (c < 3)       out[OUT_POS + (size_t)row * 3 + c] = pos_skip[(size_t)row * 3 + c];
  else if (c == 3) out[OUT_BAT + row] = (float)(row >> 13);   // row / N_DST
}

// ---------------------------------------------------------------
extern "C" void kernel_launch(void* const* d_in, const int* in_sizes, int n_in,
                              void* d_out, int out_size, void* d_ws, size_t ws_size,
                              hipStream_t stream) {
  (void)in_sizes; (void)n_in; (void)out_size; (void)ws_size;

  const float* x        = (const float*)d_in[0];
  const float* pos      = (const float*)d_in[1];
  const float* x_skip   = (const float*)d_in[3];
  const float* pos_skip = (const float*)d_in[4];
  const float* W1  = (const float*)d_in[6];
  const float* b1  = (const float*)d_in[7];
  const float* g1  = (const float*)d_in[8];
  const float* be1 = (const float*)d_in[9];
  const float* W2  = (const float*)d_in[10];
  const float* b2  = (const float*)d_in[11];
  const float* g2  = (const float*)d_in[12];
  const float* be2 = (const float*)d_in[13];

  char* ws = (char*)d_ws;
  _Float16* h0    = (_Float16*)(ws + H0_OFF);
  _Float16* h1    = (_Float16*)(ws + H1_OFF);
  float*    h2    = (float*)(ws + H2_OFF);
  _Float16* w1h   = (_Float16*)(ws + W1H_OFF);
  _Float16* w2h   = (_Float16*)(ws + W2H_OFF);
  float*    stats = (float*)(ws + STATS_OFF);   // sum1, sq1, sum2, sq2
  float*    s2    = (float*)(ws + S2_OFF);
  float*    t2    = (float*)(ws + T2_OFF);
  float*    b2p   = (float*)(ws + B2P_OFF);
  float*    out   = (float*)d_out;

  prep_kernel<<<96, 256, 0, stream>>>(W1, w1h, stats);
  knn_interp_kernel<<<256, 256, 0, stream>>>(x, pos, x_skip, pos_skip, h0);
  gemm_bn_kernel<C_IN, true><<<M_ROWS / 128, 256, 0, stream>>>(
      h0, w1h, b1, h1, (float*)nullptr, stats, stats + 128);
  fold_kernel<<<1, 128, 0, stream>>>(stats, stats + 128, g1, be1, W2, b2, w2h, b2p);
  gemm_bn_kernel<C_H, false><<<M_ROWS / 128, 256, 0, stream>>>(
      h1, w2h, b2p, (_Float16*)nullptr, h2, stats + 256, stats + 384);
  finalize2_kernel<<<1, 128, 0, stream>>>(stats + 256, stats + 384, g2, be2, s2, t2);
  epilogue_kernel<<<(int)(OUT_H / 256), 256, 0, stream>>>(h2, s2, t2, pos_skip, out);
}
